// Modified_SSDLiteMobileViT_31980326486529
// MI455X (gfx1250) — compile-verified, matrix operand order checked by reading, not run on hardware
//
#include <hip/hip_runtime.h>
#include <hip/hip_bf16.h>

#define Bn 16
#define Nn 131072
#define Mn 50
#define Cn 8
#define Pn 3
#define THREADS 256
#define GRIDX (Nn / THREADS)      // 512 blocks along N
#define NBP (Bn * GRIDX)          // 8192 partial blocks (multiple of 64 and 512)
#define IOU_THRESH 0.5f
#define NEG_POS_RATIO 3
#define PHENO_W 1e-4f

typedef __attribute__((ext_vector_type(2)))  float     v2f;
typedef __attribute__((ext_vector_type(8)))  float     v8f;
typedef __attribute__((ext_vector_type(16))) _Float16  v16h;

// ---------------- Kernel 1: IoU + per-anchor match + per-gt best anchor ----
__global__ void k_iou_match(const float4* __restrict__ anchors,
                            const float4* __restrict__ gt_boxes,
                            int* __restrict__ matched,
                            unsigned long long* __restrict__ bestgt) {
  __shared__ float4 sg[Mn];
  __shared__ float  sarea[Mn];
  __shared__ unsigned long long sbest[Mn];
  const int b = blockIdx.y;
  const int t = threadIdx.x;
  if (t < Mn) {
    float4 g = gt_boxes[b * Mn + t];
    sg[t] = g;
    sarea[t] = (g.z - g.x) * (g.w - g.y);
    sbest[t] = 0ULL;
  }
  __syncthreads();

  const int n = blockIdx.x * blockDim.x + t;
  const size_t idx = (size_t)b * Nn + n;
  const float4 a = anchors[idx];
  const float areaA = (a.z - a.x) * (a.w - a.y);

  float best = -1.0f;
  int   barg = 0;
  const unsigned lane = t & 31u;

  for (int m = 0; m < Mn; ++m) {
    float4 g = sg[m];
    float lx = fmaxf(g.x, a.x), ly = fmaxf(g.y, a.y);
    float rx = fminf(g.z, a.z), ry = fminf(g.w, a.w);
    float w = fmaxf(rx - lx, 0.0f), h = fmaxf(ry - ly, 0.0f);
    float inter = w * h;
    float iou = inter / (sarea[m] + areaA - inter);
    if (iou > best) { best = iou; barg = m; }  // first-max tie-break (ascending m, strict >)

    // per-gt argmax over anchors: pack (iou, ~anchor) so ties pick smallest anchor
    unsigned long long key =
        ((unsigned long long)__float_as_uint(iou) << 32) |
        (unsigned long long)(0xFFFFFFFFu - (unsigned)n);
    #pragma unroll
    for (int off = 16; off > 0; off >>= 1) {
      unsigned long long o = __shfl_xor(key, off, 32);
      if (o > key) key = o;
    }
    if (lane == 0) atomicMax(&sbest[m], key);
  }

  matched[idx] = (best < IOU_THRESH) ? -1 : barg;
  __syncthreads();
  if (t < Mn) atomicMax(&bestgt[b * Mn + t], sbest[t]);
}

// ---------------- Kernel 2: force-match best anchor per gt (last-gt wins) --
__global__ void k_force(const unsigned long long* __restrict__ bestgt,
                        int* __restrict__ matched) {
  int b = blockIdx.x * blockDim.x + threadIdx.x;
  if (b >= Bn) return;
  for (int m = 0; m < Mn; ++m) {
    unsigned long long k = bestgt[b * Mn + m];
    unsigned n = 0xFFFFFFFFu - (unsigned)(k & 0xFFFFFFFFu);
    if (n < Nn) matched[(size_t)b * Nn + n] = m;
  }
}

// ---------------- Kernel 3: per-anchor losses + block partials -------------
__device__ __forceinline__ float smoothl1(float d) {
  float ad = fabsf(d);
  return (ad < 1.0f) ? 0.5f * d * d : ad - 0.5f;
}

__global__ void k_losses(const float*  __restrict__ cls_logits,
                         const float4* __restrict__ bbox_reg,
                         const float*  __restrict__ ph_pred,
                         const float4* __restrict__ anchors,
                         const float4* __restrict__ gt_boxes,
                         const float*  __restrict__ gt_ph,
                         const int*    __restrict__ gt_labels,
                         const int*    __restrict__ matched,
                         float* __restrict__ negloss,
                         float* __restrict__ partials,   // SoA: 4 channels x NBP
                         int*   __restrict__ numfg) {
  const int b = blockIdx.y;
  const int t = threadIdx.x;
  const int n = blockIdx.x * blockDim.x + t;
  const size_t idx = (size_t)b * Nn + n;

  const int match = matched[idx];
  const bool fg = (match >= 0);
  const int mi = fg ? match : 0;

  // ---- bbox encode + smooth L1
  float4 a = anchors[idx];
  float4 g = gt_boxes[b * Mn + mi];
  float aw = a.z - a.x, ah = a.w - a.y;
  float ax = a.x + 0.5f * aw, ay = a.y + 0.5f * ah;
  float gw = g.z - g.x, gh = g.w - g.y;
  float gx = g.x + 0.5f * gw, gy = g.y + 0.5f * gh;
  float t0 = 10.0f * (gx - ax) / aw;
  float t1 = 10.0f * (gy - ay) / ah;
  float t2 = 5.0f * __logf(gw / aw);
  float t3 = 5.0f * __logf(gh / ah);
  float4 r = bbox_reg[idx];
  float sl1 = smoothl1(r.x - t0) + smoothl1(r.y - t1) +
              smoothl1(r.z - t2) + smoothl1(r.w - t3);

  // ---- classification cross entropy (C = 8, two float4 loads)
  const float4* lg4 = (const float4*)(cls_logits + idx * Cn);
  float4 l0 = lg4[0], l1 = lg4[1];
  float l[Cn] = {l0.x, l0.y, l0.z, l0.w, l1.x, l1.y, l1.z, l1.w};
  float mx = l[0];
  #pragma unroll
  for (int i = 1; i < Cn; ++i) mx = fmaxf(mx, l[i]);
  float se = 0.0f;
  #pragma unroll
  for (int i = 0; i < Cn; ++i) se += __expf(l[i] - mx);
  float lse = mx + __logf(se);
  int tgt = fg ? gt_labels[b * Mn + mi] : 0;
  float closs = lse - l[tgt];
  negloss[idx] = fg ? -1.0f : closs;   // fg anchors excluded from mining

  // ---- phenotype MSE
  const float* pp = ph_pred + idx * Pn;
  const float* gp = gt_ph + ((size_t)b * Mn + mi) * Pn;
  float d0 = pp[0] - gp[0], d1 = pp[1] - gp[1], d2 = pp[2] - gp[2];
  float ph = d0 * d0 + d1 * d1 + d2 * d2;

  float vsl1 = fg ? sl1 : 0.0f;
  float vcls = fg ? closs : 0.0f;
  float vph  = fg ? ph : 0.0f;
  int   vfg  = fg ? 1 : 0;

  // ---- block tree reduction
  __shared__ float s0[THREADS], s1[THREADS], s2[THREADS];
  __shared__ int   s3[THREADS];
  s0[t] = vsl1; s1[t] = vcls; s2[t] = vph; s3[t] = vfg;
  __syncthreads();
  for (int off = THREADS / 2; off > 0; off >>= 1) {
    if (t < off) {
      s0[t] += s0[t + off]; s1[t] += s1[t + off];
      s2[t] += s2[t + off]; s3[t] += s3[t + off];
    }
    __syncthreads();
  }
  if (t == 0) {
    int blk = b * GRIDX + blockIdx.x;           // 0..NBP-1
    partials[0 * NBP + blk] = s0[0];
    partials[1 * NBP + blk] = s1[0];
    partials[2 * NBP + blk] = s2[0];
    partials[3 * NBP + blk] = (float)s3[0];
    atomicAdd(&numfg[b], s3[0]);                // exact integer count
  }
}

// ---------------- Kernel 4: WMMA (matrix-pipe) reduction of partials -------
// Sum-reduction via D = A x ones + C on v_wmma_f32_16x16x4_f32.
// One wave per channel; any assignment of values into A sums them all.
// Total = sum over column N=0 of D: lane 0 holds M=0..7 (c[0..7]), lane 16 holds M=8..15.
__global__ void k_wmma_reduce(const float* __restrict__ partials,
                              float* __restrict__ totals) {
  const int wave = threadIdx.x >> 5;   // 0..3 channel
  const int lane = threadIdx.x & 31;
  const float* src = partials + wave * NBP;
  v8f c = {};

#if __has_builtin(__builtin_amdgcn_wmma_f32_16x16x4_f32)
  v2f ones; ones[0] = 1.0f; ones[1] = 1.0f;
  for (int base = 0; base < NBP; base += 64) {
    v2f a;
    a[0] = src[base + lane];
    a[1] = src[base + 32 + lane];
    c = __builtin_amdgcn_wmma_f32_16x16x4_f32(false, a, false, ones,
                                              (short)0, c, false, false);
  }
#else
  // Fallback: codegen-confirmed f16 WMMA (16x16x32), 512 values per step.
  v16h ones;
  #pragma unroll
  for (int i = 0; i < 16; ++i) ones[i] = (_Float16)1.0f;
  for (int base = 0; base < NBP; base += 512) {
    v16h a;
    #pragma unroll
    for (int j = 0; j < 16; ++j) a[j] = (_Float16)src[base + lane * 16 + j];
    c = __builtin_amdgcn_wmma_f32_16x16x32_f16(false, a, false, ones,
                                               (short)0, c, false, false);
  }
#endif

  float partial = 0.0f;
  #pragma unroll
  for (int i = 0; i < 8; ++i) partial += c[i];
  float other = __shfl_xor(partial, 16, 32);
  if (lane == 0) totals[wave] = partial + other;
}

// ---------------- Kernel 5: hard-negative mining (top-K sum per batch) -----
#define MINE_T 1024
__global__ void k_mine(const float* __restrict__ negloss,
                       const int*   __restrict__ numfg,
                       float* __restrict__ negsum) {
  const int b = blockIdx.x;
  const int t = threadIdx.x;
  const float* x = negloss + (size_t)b * Nn;
  __shared__ int   sci[MINE_T];
  __shared__ float scf[MINE_T];
  __shared__ int   bres;

  const int K = NEG_POS_RATIO * numfg[b];
  const int negc = Nn - numfg[b];

  if (K <= 0) { if (t == 0) negsum[b] = 0.0f; return; }

  if (K >= negc) {  // select all negatives
    float s = 0.0f;
    for (int i = t; i < Nn; i += MINE_T) { float v = x[i]; if (v >= 0.0f) s += v; }
    scf[t] = s; __syncthreads();
    for (int off = MINE_T / 2; off > 0; off >>= 1) {
      if (t < off) scf[t] += scf[t + off];
      __syncthreads();
    }
    if (t == 0) negsum[b] = scf[0];
    return;
  }

  // binary search for largest u with count(x >= as_float(u)) >= K
  // (negloss values are -1 for fg and >= 0 for negatives; bit-space search
  //  over nonnegative floats is monotone)
  unsigned lo = 0u, hi = 0x7F800000u;
  while (lo + 1u < hi) {
    unsigned mid = (lo + hi) >> 1;
    float fm = __uint_as_float(mid);
    int cnt = 0;
    for (int i = t; i < Nn; i += MINE_T) cnt += (x[i] >= fm) ? 1 : 0;
    sci[t] = cnt; __syncthreads();
    for (int off = MINE_T / 2; off > 0; off >>= 1) {
      if (t < off) sci[t] += sci[t + off];
      __syncthreads();
    }
    if (t == 0) bres = sci[0];
    __syncthreads();
    if (bres >= K) lo = mid; else hi = mid;
    __syncthreads();
  }

  const float v = __uint_as_float(lo);
  float s = 0.0f; int cg = 0;
  for (int i = t; i < Nn; i += MINE_T) {
    float xv = x[i];
    if (xv > v) { s += xv; ++cg; }
  }
  scf[t] = s; sci[t] = cg; __syncthreads();
  for (int off = MINE_T / 2; off > 0; off >>= 1) {
    if (t < off) { scf[t] += scf[t + off]; sci[t] += sci[t + off]; }
    __syncthreads();
  }
  if (t == 0) negsum[b] = scf[0] + (float)(K - sci[0]) * v;
}

// ---------------- Kernel 6: finalize ---------------------------------------
__global__ void k_final(const float* __restrict__ totals,
                        const float* __restrict__ negsum,
                        const int*   __restrict__ numfg,
                        float* __restrict__ out) {
  if (threadIdx.x != 0 || blockIdx.x != 0) return;
  int nf = 0;
  for (int b = 0; b < Bn; ++b) nf += numfg[b];
  float n = (float)max(nf, 1);
  float negs = 0.0f;
  for (int b = 0; b < Bn; ++b) negs += negsum[b];
  out[0] = totals[0] / n;
  out[1] = (totals[1] + negs) / n;
  out[2] = totals[2] / n * PHENO_W;
}

// ---------------- Init: zero accumulators ----------------------------------
__global__ void k_init(unsigned long long* __restrict__ bestgt,
                       int* __restrict__ numfg,
                       float* __restrict__ negsum,
                       float* __restrict__ totals) {
  int i = threadIdx.x + blockIdx.x * blockDim.x;
  if (i < Bn * Mn) bestgt[i] = 0ULL;
  if (i < Bn) { numfg[i] = 0; negsum[i] = 0.0f; }
  if (i < 4) totals[i] = 0.0f;
}

// ---------------- Host launcher --------------------------------------------
extern "C" void kernel_launch(void* const* d_in, const int* in_sizes, int n_in,
                              void* d_out, int out_size, void* d_ws, size_t ws_size,
                              hipStream_t stream) {
  const float*  cls_logits = (const float*)d_in[0];
  const float4* bbox_reg   = (const float4*)d_in[1];
  const float*  ph_pred    = (const float*)d_in[2];
  const float4* anchors    = (const float4*)d_in[3];
  const float4* gt_boxes   = (const float4*)d_in[4];
  const float*  gt_ph      = (const float*)d_in[5];
  const int*    gt_labels  = (const int*)d_in[6];
  float* out = (float*)d_out;

  char* ws = (char*)d_ws;
  size_t off = 0;
  int* matched = (int*)(ws + off);                 off += (size_t)Bn * Nn * sizeof(int);
  float* negloss = (float*)(ws + off);             off += (size_t)Bn * Nn * sizeof(float);
  float* partials = (float*)(ws + off);            off += (size_t)4 * NBP * sizeof(float);
  unsigned long long* bestgt =
      (unsigned long long*)(ws + off);             off += (size_t)Bn * Mn * sizeof(unsigned long long);
  int* numfg = (int*)(ws + off);                   off += 64;
  float* negsum = (float*)(ws + off);              off += 64;
  float* totals = (float*)(ws + off);              off += 64;

  k_init<<<dim3((Bn * Mn + 255) / 256), dim3(256), 0, stream>>>(bestgt, numfg, negsum, totals);

  dim3 grid(GRIDX, Bn);
  k_iou_match<<<grid, dim3(THREADS), 0, stream>>>(anchors, gt_boxes, matched, bestgt);
  k_force<<<dim3(1), dim3(Bn), 0, stream>>>(bestgt, matched);
  k_losses<<<grid, dim3(THREADS), 0, stream>>>(cls_logits, bbox_reg, ph_pred, anchors,
                                               gt_boxes, gt_ph, gt_labels, matched,
                                               negloss, partials, numfg);
  k_wmma_reduce<<<dim3(1), dim3(128), 0, stream>>>(partials, totals);
  k_mine<<<dim3(Bn), dim3(MINE_T), 0, stream>>>(negloss, numfg, negsum);
  k_final<<<dim3(1), dim3(1), 0, stream>>>(totals, negsum, numfg, out);
}
